// MetaLinearHeadML_45148696215988
// MI455X (gfx1250) — matrix operand order repaired and checked by reading, not compiled, not run on hardware
//
#include <hip/hip_runtime.h>

// ---------------------------------------------------------------------------
// MetaLinearHeadML for MI455X (gfx1250, wave32, WMMA bf16 16x16x32)
// ---------------------------------------------------------------------------
// S=65536, E=768, C=64, HD=256, OUT_DIM=1537
//
//   k_prep : fp32 -> bf16 weight conversion + zero accumulators (in d_ws)
//   k_main : per 128-row tile of S (512 WGs, 8 waves):
//              X chunk staged raw-f32 into LDS via GLOBAL_LOAD_ASYNC_TO_LDS
//              H1 = relu(X @ W1^T + b1)   (WMMA bf16, K=768, v_perm pack)
//              H  = relu(H1 @ W2^T + b2)  (WMMA bf16, K=256)
//              P += ys^T @ H ; colsum += sum(H) ; ysum += sum(ys)
//   k_r    : r = [P/denom_pos, (colsum-P)/denom_neg]  -> bf16 (64x512)
//   k_psi1 : A1 = relu(r @ psi_w1^T + psi_b1)    (WMMA bf16)
//   k_psi2 : params = A1 @ psi_w2^T + psi_b2     (WMMA bf16, 97 col tiles)
//   k_out  : W = W0 + sigmoid(gate)*dW ; b = b0 + db  -> d_out
// ---------------------------------------------------------------------------

typedef __attribute__((ext_vector_type(16))) __bf16          v16b;
typedef __attribute__((ext_vector_type(8)))  float           v8f;
typedef __attribute__((ext_vector_type(4)))  float           f32x4;
typedef __attribute__((ext_vector_type(4)))  int             v4i;
typedef __attribute__((ext_vector_type(8)))  unsigned short  u16x8;
typedef __attribute__((ext_vector_type(16))) unsigned short  u16x16;
typedef __attribute__((ext_vector_type(8)))  unsigned int    u32x8;

#define AS1 __attribute__((address_space(1)))
#define AS3 __attribute__((address_space(3)))

#if defined(__AMDGCN__) &&                                            \
    __has_builtin(__builtin_amdgcn_global_load_async_to_lds_b128) &&  \
    __has_builtin(__builtin_amdgcn_s_wait_asynccnt)
#define USE_ASYNC 1
#else
#define USE_ASYNC 0
#endif

#define ROWS     128
#define PITCH_XF 132   // f32 pitch: 128 + 4 pad -> 528B rows, conflict-free b128
#define PITCH_H  264   // bf16 pitch: 256 + 8 pad
#define PITCH_Y  72    // bf16 pitch: 64 + 8 pad
#define SMEM_BYTES (ROWS * PITCH_H * 2 + ROWS * PITCH_XF * 4 + ROWS * PITCH_Y * 2) // 153600

__device__ __forceinline__ unsigned short f2bf(float f) {
  unsigned int u = __builtin_bit_cast(unsigned int, f);
  u += 0x7FFFu + ((u >> 16) & 1u);          // round-to-nearest-even
  return (unsigned short)(u >> 16);
}

// Pack two f32 into packed bf16x2 with one v_perm_b32 (truncating round).
__device__ __forceinline__ unsigned int pkbf(float lo, float hi) {
  return __builtin_amdgcn_perm(__builtin_bit_cast(unsigned int, hi),
                               __builtin_bit_cast(unsigned int, lo),
                               0x07060302u);
}

// bf16 fragment from two contiguous 8x bf16 runs (two b128 loads).
__device__ __forceinline__ v16b frag_ld2(const unsigned short* p0,
                                         const unsigned short* p1) {
  u16x8 lo = *(const u16x8*)p0;
  u16x8 hi = *(const u16x8*)p1;
  u16x16 t;
#pragma unroll
  for (int i = 0; i < 8; ++i) { t[i] = lo[i]; t[i + 8] = hi[i]; }
  return __builtin_bit_cast(v16b, t);
}

// bf16 fragment converted from two contiguous 8x f32 runs (4x ds_load_b128,
// 8x v_perm_b32).
__device__ __forceinline__ v16b frag_cvt_f32(const float* p0, const float* p1) {
  f32x4 a0 = *(const f32x4*)p0;
  f32x4 a1 = *(const f32x4*)(p0 + 4);
  f32x4 b0 = *(const f32x4*)p1;
  f32x4 b1 = *(const f32x4*)(p1 + 4);
  u32x8 t;
  t[0] = pkbf(a0[0], a0[1]);
  t[1] = pkbf(a0[2], a0[3]);
  t[2] = pkbf(a1[0], a1[1]);
  t[3] = pkbf(a1[2], a1[3]);
  t[4] = pkbf(b0[0], b0[1]);
  t[5] = pkbf(b0[2], b0[3]);
  t[6] = pkbf(b1[0], b1[1]);
  t[7] = pkbf(b1[2], b1[3]);
  return __builtin_bit_cast(v16b, t);
}

__device__ __forceinline__ v8f wmma_bf16(v16b a, v16b b, v8f c) {
  return __builtin_amdgcn_wmma_f32_16x16x32_bf16(false, a, false, b,
                                                 (short)0, c, false, false);
}

// ---------------------------------------------------------------------------
// k_prep: weight fp32->bf16 + zero accumulators
// ---------------------------------------------------------------------------
__global__ void k_prep(const float* __restrict__ w1, const float* __restrict__ w2,
                       const float* __restrict__ pw1, const float* __restrict__ pw2,
                       unsigned short* __restrict__ w1bf, unsigned short* __restrict__ w2bf,
                       unsigned short* __restrict__ pw1bf, unsigned short* __restrict__ pw2bf,
                       float* __restrict__ accP, float* __restrict__ accCol,
                       float* __restrict__ accYs) {
  int i = blockIdx.x * 256 + threadIdx.x;
  int stride = gridDim.x * 256;
  for (int idx = i; idx < 256 * 768;  idx += stride) w1bf[idx]  = f2bf(w1[idx]);
  for (int idx = i; idx < 256 * 256;  idx += stride) w2bf[idx]  = f2bf(w2[idx]);
  for (int idx = i; idx < 256 * 512;  idx += stride) pw1bf[idx] = f2bf(pw1[idx]);
  for (int idx = i; idx < 1537 * 256; idx += stride) pw2bf[idx] = f2bf(pw2[idx]);
  for (int idx = i; idx < 64 * 256;   idx += stride) accP[idx]  = 0.f;
  for (int idx = i; idx < 256;        idx += stride) accCol[idx] = 0.f;
  for (int idx = i; idx < 64;         idx += stride) accYs[idx]  = 0.f;
}

// ---------------------------------------------------------------------------
// k_main: fused phi-MLP + pooled reductions over a 128-row tile of S
// ---------------------------------------------------------------------------
__global__ __launch_bounds__(256) void k_main(
    const float* __restrict__ Xs, const float* __restrict__ ys,
    const unsigned short* __restrict__ w1bf, const float* __restrict__ b1,
    const unsigned short* __restrict__ w2bf, const float* __restrict__ b2,
    float* __restrict__ accP, float* __restrict__ accCol,
    float* __restrict__ accYs) {
  extern __shared__ char smem[];
  unsigned short* sH1 = (unsigned short*)smem;                         // 128x264 bf16
  float*          sXf = (float*)(smem + ROWS * PITCH_H * 2);           // 128x132 f32
  unsigned short* sH  = (unsigned short*)(smem + ROWS * PITCH_H * 2);  // union w/ sXf
  unsigned short* sY  = (unsigned short*)(smem + ROWS * PITCH_H * 2 +
                                          ROWS * PITCH_XF * 4);        // 128x72 bf16

  const int tid  = threadIdx.x;
  const int lane = tid & 31;
  const int wv   = tid >> 5;        // wave 0..7
  const int lhi  = lane >> 4;       // upper lane half
  const int l15  = lane & 15;
  const int row0 = blockIdx.x * ROWS;
  const int mrow = 16 * wv + l15;   // A-fragment row for this lane

  const v8f zero8 = {0.f, 0.f, 0.f, 0.f, 0.f, 0.f, 0.f, 0.f};

  // Stage ys tile -> bf16 LDS, and accumulate ysum (f32 from global).
  for (int idx = tid; idx < ROWS * 64; idx += 256) {
    int r = idx >> 6, c = idx & 63;
    sY[r * PITCH_Y + c] = f2bf(ys[(size_t)(row0 + r) * 64 + c]);
  }
  if (tid < 64) {
    float s = 0.f;
    for (int r = 0; r < ROWS; ++r) s += ys[(size_t)(row0 + r) * 64 + tid];
    atomicAdd(&accYs[tid], s);
  }

  // -------------------- GEMM1: H1 = relu(X @ W1^T + b1) --------------------
  v8f acc[16];
#pragma unroll
  for (int n = 0; n < 16; ++n) acc[n] = zero8;

  for (int kc = 0; kc < 6; ++kc) {             // K chunks of 128 (E = 768)
    __syncthreads();
    // Raw f32 async DMA: 128 rows x 512B (32x b128 per row), 16 per thread.
    for (int id = tid; id < ROWS * 32; id += 256) {
      int r = id >> 5, c = id & 31;
      const float* g = &Xs[(size_t)(row0 + r) * 768 + kc * 128 + c * 4];
#if USE_ASYNC
      __builtin_amdgcn_global_load_async_to_lds_b128(
          (AS1 v4i*)g, (AS3 v4i*)&sXf[r * PITCH_XF + c * 4], 0, 0);
#else
      *(f32x4*)&sXf[r * PITCH_XF + c * 4] = *(const f32x4*)g;
#endif
      if (kc < 5) __builtin_prefetch((const void*)(g + 128), 0, 0); // next chunk
    }
#if USE_ASYNC
    __builtin_amdgcn_s_wait_asynccnt(0);
#endif
    __syncthreads();
#pragma unroll
    for (int k4 = 0; k4 < 4; ++k4) {           // K = 32 per WMMA
      int ka = k4 * 32 + (lhi ? 16 : 0);
      v16b a = frag_cvt_f32(&sXf[mrow * PITCH_XF + ka],
                            &sXf[mrow * PITCH_XF + ka + 8]);
      int kb = kc * 128 + k4 * 32 + (lhi ? 16 : 0);
      const unsigned short* bp0 = &w1bf[(size_t)l15 * 768 + kb];
      v16b bcur = frag_ld2(bp0, bp0 + 8);
#pragma unroll
      for (int n = 0; n < 16; ++n) {           // B double-buffered
        v16b bnxt = bcur;
        if (n < 15) {
          const unsigned short* bq = &w1bf[(size_t)((n + 1) * 16 + l15) * 768 + kb];
          bnxt = frag_ld2(bq, bq + 8);
        }
        acc[n] = wmma_bf16(a, bcur, acc[n]);
        bcur = bnxt;
      }
    }
  }
  // epilogue: bias + relu -> sH1 (bf16)
#pragma unroll
  for (int n = 0; n < 16; ++n) {
    int ng = n * 16 + l15;
    float bb = b1[ng];
#pragma unroll
    for (int v = 0; v < 8; ++v) {
      float x = acc[n][v] + bb;
      x = x > 0.f ? x : 0.f;
      sH1[(16 * wv + v + 8 * lhi) * PITCH_H + ng] = f2bf(x);
    }
  }
  __syncthreads();   // also: last sXf read is before this point (all waves)

  // -------------------- GEMM2: H = relu(H1 @ W2^T + b2) --------------------
#pragma unroll
  for (int n = 0; n < 16; ++n) acc[n] = zero8;
#pragma unroll
  for (int k4 = 0; k4 < 8; ++k4) {             // K = 256
    int ka = k4 * 32 + (lhi ? 16 : 0);
    v16b a = frag_ld2(&sH1[mrow * PITCH_H + ka], &sH1[mrow * PITCH_H + ka + 8]);
    int kb = k4 * 32 + (lhi ? 16 : 0);
    const unsigned short* bp0 = &w2bf[(size_t)l15 * 256 + kb];
    v16b bcur = frag_ld2(bp0, bp0 + 8);
#pragma unroll
    for (int n = 0; n < 16; ++n) {             // B double-buffered
      v16b bnxt = bcur;
      if (n < 15) {
        const unsigned short* bq = &w2bf[(size_t)((n + 1) * 16 + l15) * 256 + kb];
        bnxt = frag_ld2(bq, bq + 8);
      }
      acc[n] = wmma_bf16(a, bcur, acc[n]);
      bcur = bnxt;
    }
  }
  // epilogue: bias + relu -> sH (reuses sXf region) + column sums
#pragma unroll
  for (int n = 0; n < 16; ++n) {
    int ng = n * 16 + l15;
    float bb = b2[ng];
    float csum = 0.f;
#pragma unroll
    for (int v = 0; v < 8; ++v) {
      float x = acc[n][v] + bb;
      x = x > 0.f ? x : 0.f;
      sH[(16 * wv + v + 8 * lhi) * PITCH_H + ng] = f2bf(x);
      csum += x;
    }
    atomicAdd(&accCol[ng], csum);
  }
  __syncthreads();

  // -------------------- GEMM3: P += ys^T @ H (M=64,N=256,K=128) ------------
  {
    const int mt = wv & 3;                  // class tile (0..3)
    const int nb = (wv >> 2) * 8;           // 8 n-tiles per wave
    const int mclass = mt * 16 + l15;
    v8f acc3[8];
#pragma unroll
    for (int t = 0; t < 8; ++t) acc3[t] = zero8;
#pragma unroll
    for (int k4 = 0; k4 < 4; ++k4) {
      // A[m][k] = ys[k][m] -> strided scalar LDS reads, k = 16*lhi + e (same order as B)
      u16x16 ta;
#pragma unroll
      for (int e = 0; e < 16; ++e) {
        int kk = k4 * 32 + (lhi ? 16 : 0) + e;
        ta[e] = sY[kk * PITCH_Y + mclass];
      }
      v16b a = __builtin_bit_cast(v16b, ta);
      int kb0 = k4 * 32 + (lhi ? 16 : 0);
#pragma unroll
      for (int t = 0; t < 8; ++t) {
        int ng = (nb + t) * 16 + l15;
        u16x16 tb;
#pragma unroll
        for (int e = 0; e < 16; ++e) tb[e] = sH[(kb0 + e) * PITCH_H + ng];
        acc3[t] = wmma_bf16(a, __builtin_bit_cast(v16b, tb), acc3[t]);
      }
    }
#pragma unroll
    for (int t = 0; t < 8; ++t) {
      int ng = (nb + t) * 16 + l15;
#pragma unroll
      for (int v = 0; v < 8; ++v) {
        int m = mt * 16 + v + 8 * lhi;
        atomicAdd(&accP[m * 256 + ng], acc3[t][v]);
      }
    }
  }
}

// ---------------------------------------------------------------------------
// k_r: r = [P/denom_pos, (colsum - P)/denom_neg] -> bf16 (64 x 512)
// ---------------------------------------------------------------------------
__global__ void k_r(const float* __restrict__ accP, const float* __restrict__ accCol,
                    const float* __restrict__ accYs, unsigned short* __restrict__ rbf) {
  int idx = blockIdx.x * 256 + threadIdx.x;
  if (idx >= 64 * 512) return;
  int c = idx >> 9, n = idx & 511;
  float v;
  if (n < 256) {
    float d = accYs[c]; d = d > 1e-6f ? d : 1e-6f;
    v = accP[c * 256 + n] / d;
  } else {
    int nn = n - 256;
    float d = 65536.0f - accYs[c]; d = d > 1e-6f ? d : 1e-6f;
    v = (accCol[nn] - accP[c * 256 + nn]) / d;
  }
  rbf[idx] = f2bf(v);
}

// ---------------------------------------------------------------------------
// k_psi1: A1 = relu(r @ psi_w1^T + psi_b1) -> bf16 (64 x 256)
// grid 16 (n-tiles), 128 thr = 4 waves (m-tiles)
// ---------------------------------------------------------------------------
__global__ __launch_bounds__(128) void k_psi1(
    const unsigned short* __restrict__ rbf, const unsigned short* __restrict__ pw1bf,
    const float* __restrict__ pb1, unsigned short* __restrict__ a1bf) {
  int lane = threadIdx.x & 31, wv = threadIdx.x >> 5;
  int lhi = lane >> 4, l15 = lane & 15;
  int ng = blockIdx.x * 16 + l15;
  v8f acc = {0.f, 0.f, 0.f, 0.f, 0.f, 0.f, 0.f, 0.f};
  for (int kc = 0; kc < 16; ++kc) {            // K = 512
    int ka = kc * 32 + (lhi ? 16 : 0);
    const unsigned short* ap = &rbf[(size_t)(16 * wv + l15) * 512 + ka];
    int kb = kc * 32 + (lhi ? 16 : 0);
    const unsigned short* bp = &pw1bf[(size_t)ng * 512 + kb];
    acc = wmma_bf16(frag_ld2(ap, ap + 8), frag_ld2(bp, bp + 8), acc);
  }
  float bb = pb1[ng];
#pragma unroll
  for (int v = 0; v < 8; ++v) {
    float x = acc[v] + bb;
    x = x > 0.f ? x : 0.f;
    a1bf[(size_t)(16 * wv + v + 8 * lhi) * 256 + ng] = f2bf(x);
  }
}

// ---------------------------------------------------------------------------
// k_psi2: params = A1 @ psi_w2^T + psi_b2 -> f32 (64 x 1537)
// grid 97 (n-tiles, last partially masked), 128 thr = 4 waves
// ---------------------------------------------------------------------------
__global__ __launch_bounds__(128) void k_psi2(
    const unsigned short* __restrict__ a1bf, const unsigned short* __restrict__ pw2bf,
    const float* __restrict__ pb2, float* __restrict__ params) {
  int lane = threadIdx.x & 31, wv = threadIdx.x >> 5;
  int lhi = lane >> 4, l15 = lane & 15;
  int ng = blockIdx.x * 16 + l15;
  int ngc = ng < 1537 ? ng : 0;                // clamp for safe loads
  v8f acc = {0.f, 0.f, 0.f, 0.f, 0.f, 0.f, 0.f, 0.f};
  for (int kc = 0; kc < 8; ++kc) {             // K = 256
    int ka = kc * 32 + (lhi ? 16 : 0);
    const unsigned short* ap = &a1bf[(size_t)(16 * wv + l15) * 256 + ka];
    int kb = kc * 32 + (lhi ? 16 : 0);
    const unsigned short* bp = &pw2bf[(size_t)ngc * 256 + kb];
    acc = wmma_bf16(frag_ld2(ap, ap + 8), frag_ld2(bp, bp + 8), acc);
  }
  if (ng < 1537) {
    float bb = pb2[ng];
#pragma unroll
    for (int v = 0; v < 8; ++v)
      params[(size_t)(16 * wv + v + 8 * lhi) * 1537 + ng] = acc[v] + bb;
  }
}

// ---------------------------------------------------------------------------
// k_out: gating -> d_out = [W (64x768), b (64)]
// ---------------------------------------------------------------------------
__global__ void k_out(const float* __restrict__ params, const float* __restrict__ W0,
                      const float* __restrict__ b0, float* __restrict__ out) {
  int idx = blockIdx.x * 256 + threadIdx.x;
  const int NW = 64 * 768;
  if (idx < NW) {
    int c = idx / 768, j = idx % 768;
    float dW = params[(size_t)c * 1537 + j];
    float g  = params[(size_t)c * 1537 + 769 + j];
    g = 1.0f / (1.0f + __expf(-g));
    out[idx] = W0[j] + g * dW;
  } else if (idx < NW + 64) {
    int c = idx - NW;
    out[idx] = b0[0] + params[(size_t)c * 1537 + 768];
  }
}

// ---------------------------------------------------------------------------
extern "C" void kernel_launch(void* const* d_in, const int* in_sizes, int n_in,
                              void* d_out, int out_size, void* d_ws, size_t ws_size,
                              hipStream_t stream) {
  const float* Xs  = (const float*)d_in[0];
  const float* ys  = (const float*)d_in[1];
  const float* w1  = (const float*)d_in[2];
  const float* b1  = (const float*)d_in[3];
  const float* w2  = (const float*)d_in[4];
  const float* b2  = (const float*)d_in[5];
  const float* pw1 = (const float*)d_in[6];
  const float* pb1 = (const float*)d_in[7];
  const float* pw2 = (const float*)d_in[8];
  const float* pb2 = (const float*)d_in[9];
  const float* W0  = (const float*)d_in[10];
  const float* b0  = (const float*)d_in[11];

  char* ws = (char*)d_ws;
  float*          accP   = (float*)(ws + 0);         // 64*256*4   = 65536
  float*          accCol = (float*)(ws + 65536);     // 256*4      = 1024
  float*          accYs  = (float*)(ws + 66560);     // 64*4       = 256
  float*          params = (float*)(ws + 66816);     // 64*1537*4  = 393472
  unsigned short* w1bf   = (unsigned short*)(ws + 460288);   // 256*768*2
  unsigned short* w2bf   = (unsigned short*)(ws + 853504);   // 256*256*2
  unsigned short* pw1bf  = (unsigned short*)(ws + 984576);   // 256*512*2
  unsigned short* pw2bf  = (unsigned short*)(ws + 1246720);  // 1537*256*2
  unsigned short* rbf    = (unsigned short*)(ws + 2033664);  // 64*512*2
  unsigned short* a1bf   = (unsigned short*)(ws + 2099200);  // 64*256*2

  k_prep<<<512, 256, 0, stream>>>(w1, w2, pw1, pw2, w1bf, w2bf, pw1bf, pw2bf,
                                  accP, accCol, accYs);
  k_main<<<65536 / ROWS, 256, SMEM_BYTES, stream>>>(Xs, ys, w1bf, b1, w2bf, b2,
                                                    accP, accCol, accYs);
  k_r<<<(64 * 512 + 255) / 256, 256, 0, stream>>>(accP, accCol, accYs, rbf);
  k_psi1<<<16, 128, 0, stream>>>(rbf, pw1bf, pb1, a1bf);
  k_psi2<<<97, 128, 0, stream>>>(a1bf, pw2bf, pb2, params);
  k_out<<<(64 * 768 + 64 + 255) / 256, 256, 0, stream>>>(params, W0, b0,
                                                         (float*)d_out);
}
